// Het_Aggregate_10548439679672
// MI455X (gfx1250) — matrix operand order repaired and checked by reading, not compiled, not run on hardware
//
#include <hip/hip_runtime.h>

// ---------------------------------------------------------------------------
// MI455X / gfx1250: bf16 WMMA bi-LSTM content encoder + neighbor reducers +
// attention fusion.  wave32, V_WMMA_F32_16X16X32_BF16 throughout.
// Weights are staged per-block into LDS (padded rows -> conflict-free
// ds_load_b128 B-fragments); a full weight set (192KB) exceeds the 64KB WGP$,
// so feeding WMMA from LDS instead of L2 removes the dominant traffic.
// Cell state / h-sum live in per-wave LDS so the gate-tile loop stays rolled
// (prevents LICM from hoisting 96 weight fragments -> no VGPR spills).
// ---------------------------------------------------------------------------

typedef __attribute__((ext_vector_type(16))) __bf16 v16bf;
typedef __attribute__((ext_vector_type(8)))  __bf16 v8bf;
typedef __attribute__((ext_vector_type(8)))  float  v8f;

#define NN   30000
#define DD   128
#define HH   64
#define G4   256     // 4*H
#define NTILES 1875  // NN/16
#define WAVES 8      // waves per block
#define NT8  235     // ceil(NTILES/8)

#define WIH_S 136    // padded LDS row stride (bf16) for Wih rows of 128
#define WHH_S 72     // padded LDS row stride (bf16) for Whh rows of 64
#define HST_S 72     // padded LDS row stride (bf16) for h stage rows of 64

// ---------------- fragment helpers (wave32 CDNA5 WMMA layouts) -------------

__device__ __forceinline__ v16bf cat8(v8bf lo, v8bf hi) {
  v16bf r;
#pragma unroll
  for (int i = 0; i < 8; ++i) { r[i] = lo[i]; r[8 + i] = hi[i]; }
  return r;
}

// A fragment 16x32 (MxK) bf16 from row-major source; p = &row_m[kchunk*32].
// Lane L: m = L&15, kb = (L<16)?0:8; elements K=kb..kb+7 and K=kb+16..kb+23.
__device__ __forceinline__ v16bf a_frag_bf16(const __bf16* p, int lane) {
  const int kb = (lane < 16) ? 0 : 8;
  v8bf lo = *(const v8bf*)(p + kb);
  v8bf hi = *(const v8bf*)(p + kb + 16);
  return cat8(lo, hi);
}

__device__ __forceinline__ v16bf a_frag_f32(const float* p, int lane) {
  const int kb = (lane < 16) ? 0 : 8;
  v8f lo = *(const v8f*)(p + kb);
  v8f hi = *(const v8f*)(p + kb + 16);
  v16bf r;
#pragma unroll
  for (int i = 0; i < 8; ++i) { r[i] = (__bf16)lo[i]; r[8 + i] = (__bf16)hi[i]; }
  return r;
}

// B fragment 32x16 (KxN) where B[k][n] = W[n][k], W row-major (LDS, padded).
// wrow = &W[n][kchunk*32] with n = ntile*16 + (lane&15).
// Lane L: kb = (L<16)?0:8; elements K=kb..kb+7 and K=kb+16..kb+23 (matches A).
__device__ __forceinline__ v16bf b_frag(const __bf16* wrow, int lane) {
  const int kb = (lane < 16) ? 0 : 8;
  v8bf lo = *(const v8bf*)(wrow + kb);
  v8bf hi = *(const v8bf*)(wrow + kb + 16);
  return cat8(lo, hi);
}

__device__ __forceinline__ float sigmoidf_(float x) {
  return 1.0f / (1.0f + __expf(-x));
}
__device__ __forceinline__ float tanhf_(float x) {
  x = fminf(fmaxf(x, -15.0f), 15.0f);
  float e = __expf(2.0f * x);
  return (e - 1.0f) / (e + 1.0f);
}

// ---------------- input-row loaders ----------------------------------------

struct LoadX {                // content: f32 rows of x [N][3][128]
  const float* base;
  int n0;
  __device__ __forceinline__ v16bf frag(int t, int kc, int lane) const {
    int m = lane & 15;
    const float* p = base + ((size_t)(n0 + m) * 3 + t) * DD + kc * 32;
    return a_frag_f32(p, lane);
  }
  __device__ __forceinline__ void prefetch(int t, int lane) const {
    int m = lane & 15;
    __builtin_prefetch(base + ((size_t)(n0 + m) * 3 + t) * DD, 0, 1);
  }
};

struct LoadG {                // neighbor: gathered bf16 content rows
  const __bf16* base;         // content + s*N*128
  const int* idx;             // &nbr[s][d][n0][0], row stride K=10
  __device__ __forceinline__ v16bf frag(int t, int kc, int lane) const {
    int m = lane & 15;
    int r = idx[m * 10 + t];
    return a_frag_bf16(base + (size_t)r * DD + kc * 32, lane);
  }
  __device__ __forceinline__ void prefetch(int t, int lane) const {
    int m = lane & 15;
    int r = idx[m * 10 + t];
    __builtin_prefetch(base + (size_t)r * DD, 0, 1);
  }
};

// ---------------- weight staging into LDS ----------------------------------

__device__ __forceinline__ void stage_weights(const __bf16* __restrict__ Wih,
                                              const __bf16* __restrict__ Whh,
                                              const float* __restrict__ bias,
                                              __bf16* wihS, __bf16* whhS,
                                              float* biasS, int tid) {
  // Wih: 256 rows x 16 chunks of 8 bf16 -> padded stride WIH_S
#pragma unroll 1
  for (int i = tid; i < 256 * 16; i += 256) {
    int r = i >> 4, ch = i & 15;
    ((v8bf*)(wihS + r * WIH_S))[ch] = ((const v8bf*)(Wih + r * DD))[ch];
  }
  // Whh: 256 rows x 8 chunks of 8 bf16 -> padded stride WHH_S
#pragma unroll 1
  for (int i = tid; i < 256 * 8; i += 256) {
    int r = i >> 3, ch = i & 7;
    ((v8bf*)(whhS + r * WHH_S))[ch] = ((const v8bf*)(Whh + r * HH))[ch];
  }
  if (tid < 256) biasS[tid] = bias[tid];
}

// ---------------- core: one direction of a 16-node LSTM --------------------
// Mean over the sequence is order invariant, so the backward direction just
// runs the recurrence with t descending and accumulates into the same mean.

template <int T, class Loader>
__device__ void lstm_dir_mean(const Loader& ld, int dir,
                              const __bf16* wihS,  // LDS [256][WIH_S]
                              const __bf16* whhS,  // LDS [256][WHH_S]
                              const float* biasS,  // LDS [256]
                              __bf16* hstage,      // LDS [16][HST_S]
                              float* cls,          // LDS [16][64] cell state
                              float* hsls,         // LDS [16][64] h running sum
                              float* __restrict__ outF,   // [16][128] or null
                              __bf16* __restrict__ outB,  // [16][128] or null
                              int lane) {
  const int ncol = lane & 15;
  const int roff = (lane < 16) ? 0 : 8;

  // zero h stage (h0 = 0), cell state, h-sum
  {
    v8bf zb;
#pragma unroll
    for (int i = 0; i < 8; ++i) zb[i] = (__bf16)0.0f;
#pragma unroll 1
    for (int i = lane; i < 16 * HST_S / 8; i += 32) ((v8bf*)hstage)[i] = zb;
    float4 zf = make_float4(0.f, 0.f, 0.f, 0.f);
#pragma unroll
    for (int i = 0; i < 8; ++i) {
      ((float4*)cls)[lane * 8 + i] = zf;
      ((float4*)hsls)[lane * 8 + i] = zf;
    }
  }

#pragma unroll 1
  for (int st = 0; st < T; ++st) {
    const int t = dir ? (T - 1 - st) : st;
    if (st + 1 < T) ld.prefetch(dir ? (T - 2 - st) : (st + 1), lane);

    v16bf ax[4];
#pragma unroll
    for (int kc = 0; kc < 4; ++kc) ax[kc] = ld.frag(t, kc, lane);
    v16bf ah0 = a_frag_bf16(hstage + (lane & 15) * HST_S, lane);
    v16bf ah1 = a_frag_bf16(hstage + (lane & 15) * HST_S + 32, lane);

#pragma unroll 1
    for (int q = 0; q < 4; ++q) {      // column tile within H (rolled!)
      v8f acc[4];
#pragma unroll
      for (int p = 0; p < 4; ++p) {    // gate: i, f, g, o
        const int n = (p * 4 + q) * 16 + ncol;
        const float b = biasS[n];
        v8f a;
#pragma unroll
        for (int j = 0; j < 8; ++j) a[j] = b;
        const __bf16* wr = wihS + n * WIH_S;
#pragma unroll
        for (int kc = 0; kc < 4; ++kc)
          a = __builtin_amdgcn_wmma_f32_16x16x32_bf16(
              false, ax[kc], false, b_frag(wr + kc * 32, lane),
              (short)0, a, false, false);
        const __bf16* hr = whhS + n * WHH_S;
        a = __builtin_amdgcn_wmma_f32_16x16x32_bf16(
            false, ah0, false, b_frag(hr, lane), (short)0, a, false, false);
        a = __builtin_amdgcn_wmma_f32_16x16x32_bf16(
            false, ah1, false, b_frag(hr + 32, lane), (short)0, a, false, false);
        acc[p] = a;
      }
      // elementwise LSTM cell update; state lives in LDS
#pragma unroll
      for (int j = 0; j < 8; ++j) {
        const int o = (j + roff) * HH + q * 16 + ncol;
        float iv = sigmoidf_(acc[0][j]);
        float fv = sigmoidf_(acc[1][j]);
        float gv = tanhf_(acc[2][j]);
        float ov = sigmoidf_(acc[3][j]);
        float cn = fmaf(fv, cls[o], iv * gv);
        cls[o] = cn;
        float hn = ov * tanhf_(cn);
        hsls[o] += hn;
        hstage[(j + roff) * HST_S + q * 16 + ncol] = (__bf16)hn;
      }
    }
  }

  // emit this direction's mean (cols dir*64 .. dir*64+63)
  const float inv = 1.0f / (float)T;
#pragma unroll 1
  for (int q = 0; q < 4; ++q) {
#pragma unroll
    for (int j = 0; j < 8; ++j) {
      const int row = j + roff;
      float v = hsls[row * HH + q * 16 + ncol] * inv;
      const int colc = dir * HH + q * 16 + ncol;
      if (outF) outF[(size_t)row * DD + colc] = v;
      if (outB) outB[(size_t)row * DD + colc] = (__bf16)v;
    }
  }
}

// ---------------- kernels ---------------------------------------------------

__global__ __launch_bounds__(256) void het_content_kernel(
    const float* __restrict__ xa, const float* __restrict__ xb,
    const float* __restrict__ xc, const __bf16* __restrict__ wih,
    const __bf16* __restrict__ whh, const float* __restrict__ bias,
    __bf16* __restrict__ content) {
  __shared__ __attribute__((aligned(16))) __bf16 wihS[256 * WIH_S];
  __shared__ __attribute__((aligned(16))) __bf16 whhS[256 * WHH_S];
  __shared__ __attribute__((aligned(16))) float  biasS[256];
  __shared__ __attribute__((aligned(16))) __bf16 hst[WAVES][16 * HST_S];
  __shared__ __attribute__((aligned(16))) float  cst[WAVES][16 * HH];
  __shared__ __attribute__((aligned(16))) float  hss[WAVES][16 * HH];
  const int wave = threadIdx.x >> 5, lane = threadIdx.x & 31;
  const int tile = blockIdx.x * WAVES + wave;
  const bool active = tile < NTILES;
  const int s = blockIdx.y;
  const float* x = (s == 0) ? xa : (s == 1) ? xb : xc;
  LoadX ld{x, tile * 16};
  __bf16* outB = content + ((size_t)s * NN + (size_t)tile * 16) * DD;

#pragma unroll 1
  for (int dir = 0; dir < 2; ++dir) {
    __syncthreads();
    stage_weights(wih + (size_t)dir * G4 * DD, whh + (size_t)dir * G4 * HH,
                  bias + dir * G4, wihS, whhS, biasS, threadIdx.x);
    __syncthreads();
    if (active)
      lstm_dir_mean<3>(ld, dir, wihS, whhS, biasS, hst[wave], cst[wave],
                       hss[wave], (float*)nullptr, outB, lane);
  }
}

__global__ __launch_bounds__(256) void het_neigh_kernel(
    const __bf16* __restrict__ content, const int* __restrict__ nbr,
    const __bf16* __restrict__ wihn, const __bf16* __restrict__ whhn,
    const float* __restrict__ biasn, float* __restrict__ neigh) {
  __shared__ __attribute__((aligned(16))) __bf16 wihS[256 * WIH_S];
  __shared__ __attribute__((aligned(16))) __bf16 whhS[256 * WHH_S];
  __shared__ __attribute__((aligned(16))) float  biasS[256];
  __shared__ __attribute__((aligned(16))) __bf16 hst[WAVES][16 * HST_S];
  __shared__ __attribute__((aligned(16))) float  cst[WAVES][16 * HH];
  __shared__ __attribute__((aligned(16))) float  hss[WAVES][16 * HH];
  const int wave = threadIdx.x >> 5, lane = threadIdx.x & 31;
  const int sd = blockIdx.y;         // 0..8
  const int s = sd / 3, d = sd - s * 3;
  const int tile = blockIdx.x * WAVES + wave;
  const bool active = tile < NTILES;
  const int n0 = tile * 16;
  LoadG ld{content + (size_t)s * NN * DD,
           nbr + (((size_t)s * 3 + d) * NN + n0) * 10};
  float* outF = neigh + (((size_t)d * 3 + s) * NN + n0) * DD;

#pragma unroll 1
  for (int dir = 0; dir < 2; ++dir) {
    __syncthreads();
    stage_weights(wihn + ((size_t)s * 2 + dir) * G4 * DD,
                  whhn + ((size_t)s * 2 + dir) * G4 * HH,
                  biasn + (s * 2 + dir) * G4, wihS, whhS, biasS, threadIdx.x);
    __syncthreads();
    if (active)
      lstm_dir_mean<10>(ld, dir, wihS, whhS, biasS, hst[wave], cst[wave],
                        hss[wave], outF, (__bf16*)nullptr, lane);
  }
}

__global__ __launch_bounds__(256) void het_attn_kernel(
    const __bf16* __restrict__ content, const float* __restrict__ neigh,
    const float* __restrict__ attn_w, const float* __restrict__ attn_b,
    float* __restrict__ out) {
  const int wave = threadIdx.x >> 5, lane = threadIdx.x & 31;
  const long task = (long)blockIdx.x * 8 + wave;
  if (task >= 3L * NN) return;
  const int d = (int)(task / NN);
  const int n = (int)(task - (long)d * NN);
  const int c = lane * 4;

  float dh[4];
  {
    const __bf16* p = content + ((size_t)d * NN + n) * DD + c;
#pragma unroll
    for (int i = 0; i < 4; ++i) dh[i] = (float)p[i];
  }
  float e[4][4];
#pragma unroll
  for (int k = 0; k < 3; ++k) {
    const float* p = neigh + (((size_t)d * 3 + k) * NN + n) * DD + c;
#pragma unroll
    for (int i = 0; i < 4; ++i) e[k][i] = p[i];
  }
#pragma unroll
  for (int i = 0; i < 4; ++i) e[3][i] = dh[i];

  const float* w1 = attn_w + d * 2 * DD + c;
  const float* w2 = attn_w + d * 2 * DD + DD + c;
  float p0 = 0.f, pk[4] = {0.f, 0.f, 0.f, 0.f};
#pragma unroll
  for (int i = 0; i < 4; ++i) {
    p0 = fmaf(w1[i], dh[i], p0);
#pragma unroll
    for (int k = 0; k < 4; ++k) pk[k] = fmaf(w2[i], e[k][i], pk[k]);
  }
#pragma unroll
  for (int m = 16; m; m >>= 1) {
    p0 += __shfl_xor(p0, m, 32);
#pragma unroll
    for (int k = 0; k < 4; ++k) pk[k] += __shfl_xor(pk[k], m, 32);
  }
  const float b = attn_b[d];
  float lg[4], mx = -1e30f;
#pragma unroll
  for (int k = 0; k < 4; ++k) {
    float l = p0 + pk[k] + b;
    l = (l > 0.f) ? l : 0.01f * l;           // leaky_relu(0.01)
    lg[k] = l;
    mx = fmaxf(mx, l);
  }
  float sum = 0.f;
#pragma unroll
  for (int k = 0; k < 4; ++k) { lg[k] = __expf(lg[k] - mx); sum += lg[k]; }
  const float inv = 1.0f / sum;
  float o[4] = {0.f, 0.f, 0.f, 0.f};
#pragma unroll
  for (int k = 0; k < 4; ++k) {
    float w = lg[k] * inv;
#pragma unroll
    for (int i = 0; i < 4; ++i) o[i] = fmaf(w, e[k][i], o[i]);
  }
  float* po = out + ((size_t)d * NN + n) * DD + c;
#pragma unroll
  for (int i = 0; i < 4; ++i) po[i] = o[i];
}

// ---------------- prep kernels ----------------------------------------------

__global__ void het_cvt_bf16_kernel(const float* __restrict__ src,
                                    __bf16* __restrict__ dst, int n) {
  int i = blockIdx.x * blockDim.x + threadIdx.x;
  if (i < n) dst[i] = (__bf16)src[i];
}
__global__ void het_bias_sum_kernel(const float* __restrict__ a,
                                    const float* __restrict__ b,
                                    float* __restrict__ o, int n) {
  int i = blockIdx.x * blockDim.x + threadIdx.x;
  if (i < n) o[i] = a[i] + b[i];
}

// ---------------- launch -----------------------------------------------------

extern "C" void kernel_launch(void* const* d_in, const int* in_sizes, int n_in,
                              void* d_out, int out_size, void* d_ws,
                              size_t ws_size, hipStream_t stream) {
  const float* xa    = (const float*)d_in[0];
  const float* xb    = (const float*)d_in[1];
  const float* xc    = (const float*)d_in[2];
  const float* Wih_c = (const float*)d_in[3];   // [2,256,128]
  const float* Whh_c = (const float*)d_in[4];   // [2,256,64]
  const float* bih_c = (const float*)d_in[5];
  const float* bhh_c = (const float*)d_in[6];
  const float* Wih_n = (const float*)d_in[7];   // [3,2,256,128]
  const float* Whh_n = (const float*)d_in[8];   // [3,2,256,64]
  const float* bih_n = (const float*)d_in[9];
  const float* bhh_n = (const float*)d_in[10];
  const float* attn_w = (const float*)d_in[11];
  const float* attn_b = (const float*)d_in[12];
  const int*   nbr    = (const int*)d_in[13];
  float* out = (float*)d_out;

  // workspace carve-up (~162 MB): content bf16, neigh f32, bf16 weights, bias
  char* ws = (char*)d_ws;
  size_t off = 0;
  auto carve = [&](size_t bytes) {
    size_t o = off;
    off += (bytes + 255) & ~(size_t)255;
    return o;
  };
  __bf16* content = (__bf16*)(ws + carve((size_t)3 * NN * DD * 2));
  float*  neigh   = (float*)(ws + carve((size_t)9 * NN * DD * 4));
  __bf16* wihc    = (__bf16*)(ws + carve((size_t)2 * G4 * DD * 2));
  __bf16* whhc    = (__bf16*)(ws + carve((size_t)2 * G4 * HH * 2));
  __bf16* wihn    = (__bf16*)(ws + carve((size_t)6 * G4 * DD * 2));
  __bf16* whhn    = (__bf16*)(ws + carve((size_t)6 * G4 * HH * 2));
  float*  biasc   = (float*)(ws + carve((size_t)2 * G4 * 4));
  float*  biasn   = (float*)(ws + carve((size_t)6 * G4 * 4));

  auto cvt = [&](const float* s, __bf16* d, int n) {
    het_cvt_bf16_kernel<<<(n + 255) / 256, 256, 0, stream>>>(s, d, n);
  };
  cvt(Wih_c, wihc, 2 * G4 * DD);
  cvt(Whh_c, whhc, 2 * G4 * HH);
  cvt(Wih_n, wihn, 6 * G4 * DD);
  cvt(Whh_n, whhn, 6 * G4 * HH);
  het_bias_sum_kernel<<<(2 * G4 + 255) / 256, 256, 0, stream>>>(
      bih_c, bhh_c, biasc, 2 * G4);
  het_bias_sum_kernel<<<(6 * G4 + 255) / 256, 256, 0, stream>>>(
      bih_n, bhh_n, biasn, 6 * G4);

  het_content_kernel<<<dim3(NT8, 3), 256, 0, stream>>>(xa, xb, xc, wihc, whhc,
                                                       biasc, content);

  het_neigh_kernel<<<dim3(NT8, 9), 256, 0, stream>>>(content, nbr, wihn, whhn,
                                                     biasn, neigh);

  long tasks3 = 3L * NN;
  het_attn_kernel<<<(int)((tasks3 + 7) / 8), 256, 0, stream>>>(
      content, neigh, attn_w, attn_b, out);
}
